// CrossAttention_12223476924600
// MI455X (gfx1250) — matrix operand order repaired and checked by reading, not compiled, not run on hardware
//
#include <hip/hip_runtime.h>
#include <hip/hip_bf16.h>
#include <stdint.h>

// ---------------------------------------------------------------------------
// Cross-attention for MI455X (gfx1250): all GEMMs via v_wmma_f32_16x16x32_f16,
// attention K/V staged through LDS with global_load_async_to_lds_b128.
// ---------------------------------------------------------------------------

#define B_      4
#define N_      2048
#define M_      2048
#define DIMX    512
#define HEADS_  8
#define DHEAD   64
#define INNER_  512
#define MK      2049      // keys incl. null token
#define KPAD    2080      // padded key count (multiple of 32)
#define WPB     4         // waves per block

typedef _Float16 v16h __attribute__((ext_vector_type(16)));
typedef _Float16 v8h  __attribute__((ext_vector_type(8)));
typedef float    v8f  __attribute__((ext_vector_type(8)));

union V16H { v16h v; v8h h8[2]; _Float16 e[16]; };

__device__ __forceinline__ v8f wmma16(v16h a, v16h b, v8f c) {
  return __builtin_amdgcn_wmma_f32_16x16x32_f16(false, a, false, b, (short)0, c,
                                                false, false);
}

// A-matrix fragment (16x32, f16) from an f32 row-major source.
// Layout (ISA 7.12.2): lane<16 holds row M=lane, K = {k0..k0+7, k0+16..k0+23};
// lane>=16 holds row M=lane-16, K = {k0+8..k0+15, k0+24..k0+31}.
__device__ __forceinline__ v16h load_a_f32(const float* A, int lda, int row0,
                                           int k0, int lane) {
  int l = lane & 15, hi = lane >> 4;
  const float* p = A + (size_t)(row0 + l) * lda + k0 + hi * 8;
  V16H a;
#pragma unroll
  for (int i = 0; i < 8; ++i) {
    a.e[i]     = (_Float16)p[i];
    a.e[8 + i] = (_Float16)p[i + 16];
  }
  return a.v;
}

// A-matrix fragment (16x32, f16) from f16 row-major source (global or LDS).
__device__ __forceinline__ v16h load_a_f16(const _Float16* A, int lda, int row0,
                                           int k0, int lane) {
  int l = lane & 15, hi = lane >> 4;
  const _Float16* p = A + (size_t)(row0 + l) * lda + k0 + hi * 8;
  V16H a;
  a.h8[0] = *(const v8h*)p;
  a.h8[1] = *(const v8h*)(p + 16);
  return a.v;
}

// B-matrix fragment (32x16, f16) from a column-contiguous source:
// BT is row-major [Ncols][ldb] where a "row" is one B-column (K contiguous).
// Layout: lane<16 -> column N=col0+lane, K={k0..k0+7, k0+16..k0+23}; lane>=16 -> K={k0+8..k0+15, k0+24..k0+31}.
__device__ __forceinline__ v16h load_b_t(const _Float16* BT, int ldb, int col0,
                                         int k0, int lane) {
  int l = lane & 15, hi = lane >> 4;
  const _Float16* p = BT + (size_t)(col0 + l) * ldb + k0 + hi * 8;
  V16H b;
  b.h8[0] = *(const v8h*)p;
  b.h8[1] = *(const v8h*)(p + 16);
  return b.v;
}

// Async 16-byte global->LDS copy (CDNA5, tracked by ASYNCcnt).
__device__ __forceinline__ void async_g2l_b128(uint32_t lds_addr,
                                               const void* gaddr) {
  uint64_t ga = (uint64_t)(uintptr_t)gaddr;
  asm volatile("global_load_async_to_lds_b128 %0, %1, off"
               :: "v"(lds_addr), "v"(ga) : "memory");
}

// ---------------------------------------------------------------------------
// Kernel 0: transpose + convert f32 weights [K][Ncol] -> f16 [Ncol][K]
// ---------------------------------------------------------------------------
__global__ void xattn_w_to_f16T(const float* __restrict__ src,
                                _Float16* __restrict__ dst, int K, int Ncol) {
  int i = blockIdx.x * blockDim.x + threadIdx.x;
  if (i >= K * Ncol) return;
  int n = i / K, k = i % K;
  dst[i] = (_Float16)src[(size_t)k * Ncol + n];
}

// ---------------------------------------------------------------------------
// Kernel 1: q = tanh(x @ Wq)  -> qh [B][H][N][64] f16
// ---------------------------------------------------------------------------
__global__ void xattn_proj_q(const float* __restrict__ x,
                             const _Float16* __restrict__ WqT,
                             _Float16* __restrict__ qh) {
  int wave = threadIdx.x >> 5, lane = threadIdx.x & 31;
  int tile = blockIdx.x * WPB + wave;            // (B*N/16) * (INNER/16) tiles
  const int tilesN = INNER_ / 16;                // 32
  int row0 = (tile / tilesN) * 16;
  int col0 = (tile % tilesN) * 16;
  v8f c = {};
#pragma unroll 4
  for (int k0 = 0; k0 < DIMX; k0 += 32) {
    v16h a = load_a_f32(x, DIMX, row0, k0, lane);
    v16h b = load_b_t(WqT, DIMX, col0, k0, lane);
    c = wmma16(a, b, c);
  }
  int l = lane & 15, hi = lane >> 4;
  int col = col0 + l;
  int h = col >> 6, d = col & 63;
#pragma unroll
  for (int r = 0; r < 8; ++r) {
    int grow = row0 + hi * 8 + r;                // global row in [0, B*N)
    int b = grow / N_, n = grow % N_;
    qh[(((size_t)b * HEADS_ + h) * N_ + n) * DHEAD + d] = (_Float16)tanhf(c[r]);
  }
}

// ---------------------------------------------------------------------------
// Kernel 2: kv = context @ Wkv; k -> tanh f16 [B][H][KPAD][64] (rows 1..2048),
//           v -> f16 transposed [B][H][64][KPAD]
// ---------------------------------------------------------------------------
__global__ void xattn_proj_kv(const float* __restrict__ ctx,
                              const _Float16* __restrict__ WkvT,
                              _Float16* __restrict__ kh,
                              _Float16* __restrict__ vT) {
  int wave = threadIdx.x >> 5, lane = threadIdx.x & 31;
  int tile = blockIdx.x * WPB + wave;            // (B*M/16) * (1024/16) tiles
  const int tilesN = (2 * INNER_) / 16;          // 64
  int row0 = (tile / tilesN) * 16;
  int col0 = (tile % tilesN) * 16;
  v8f c = {};
#pragma unroll 4
  for (int k0 = 0; k0 < DIMX; k0 += 32) {
    v16h a = load_a_f32(ctx, DIMX, row0, k0, lane);
    v16h b = load_b_t(WkvT, DIMX, col0, k0, lane);
    c = wmma16(a, b, c);
  }
  int l = lane & 15, hi = lane >> 4;
  int col = col0 + l;
#pragma unroll
  for (int r = 0; r < 8; ++r) {
    int grow = row0 + hi * 8 + r;
    int b = grow / M_, m = grow % M_;
    if (col < INNER_) {                          // k half (tanh)
      int h = col >> 6, d = col & 63;
      kh[(((size_t)b * HEADS_ + h) * KPAD + (m + 1)) * DHEAD + d] =
          (_Float16)tanhf(c[r]);
    } else {                                     // v half (transposed store)
      int c2 = col - INNER_;
      int h = c2 >> 6, d = c2 & 63;
      vT[(((size_t)b * HEADS_ + h) * DHEAD + d) * KPAD + (m + 1)] =
          (_Float16)c[r];
    }
  }
}

// ---------------------------------------------------------------------------
// Kernel 2b: null token row (j=0) + zero the padded key rows (j in [MK,KPAD))
// ---------------------------------------------------------------------------
__global__ void xattn_null_pad(const float* __restrict__ null_k,
                               const float* __restrict__ null_v,
                               _Float16* __restrict__ kh,
                               _Float16* __restrict__ vT) {
  int bh = blockIdx.x;                           // B*H blocks
  int d = threadIdx.x;                           // 64 threads
  kh[((size_t)bh * KPAD + 0) * DHEAD + d] = (_Float16)tanhf(null_k[d]);
  vT[((size_t)bh * DHEAD + d) * KPAD + 0] = (_Float16)null_v[d];
  for (int j = MK; j < KPAD; ++j) {
    kh[((size_t)bh * KPAD + j) * DHEAD + d] = (_Float16)0.f;
    vT[((size_t)bh * DHEAD + d) * KPAD + j] = (_Float16)0.f;
  }
}

// ---------------------------------------------------------------------------
// Kernel 3: flash attention. One wave per 16-row q tile; the 4 waves of a
// block share one (b,h), so K/V tiles are staged once per block into LDS via
// async global->LDS copies (ASYNCcnt) and consumed as ds_load fragments.
// ---------------------------------------------------------------------------
__global__ void xattn_attn(const _Float16* __restrict__ qh,
                           const _Float16* __restrict__ kh,
                           const _Float16* __restrict__ vT,
                           _Float16* __restrict__ ao) {
  __shared__ __align__(16) _Float16 ldsK[32 * DHEAD];     // [key][d]   4 KB
  __shared__ __align__(16) _Float16 ldsV[DHEAD * 32];     // [d][key]   4 KB
  __shared__ __align__(32) _Float16 pbuf[WPB][16 * 32];   //            4 KB
  int wave = threadIdx.x >> 5, lane = threadIdx.x & 31;
  int t = threadIdx.x;                           // 0..127
  int l = lane & 15, hi = lane >> 4;
  int tile = blockIdx.x * WPB + wave;            // B*H*(N/16) tiles
  const int ntiles = N_ / 16;                    // 128 (multiple of WPB)
  int bh = tile / ntiles;                        // same for all 4 waves
  int row0 = (tile % ntiles) * 16;
  const _Float16* Q = qh + (size_t)bh * N_ * DHEAD;
  const _Float16* K = kh + (size_t)bh * KPAD * DHEAD;
  const _Float16* V = vT + (size_t)bh * DHEAD * KPAD;
  uint32_t lk = (uint32_t)(uintptr_t)&ldsK[0];
  uint32_t lv = (uint32_t)(uintptr_t)&ldsV[0];

  // Q tile (16x64) as two A fragments, K-dim = head dim.
  v16h a0 = load_a_f16(Q, DHEAD, row0, 0, lane);
  v16h a1 = load_a_f16(Q, DHEAD, row0, 32, lane);

  v8f acc[4] = {};                               // O tile 16x64 (4 N-segments)
  float m_i[8], l_i[8];
#pragma unroll
  for (int r = 0; r < 8; ++r) { m_i[r] = -1e30f; l_i[r] = 0.f; }
  const float scale = 0.125f;                    // 64^-0.5

  for (int j0 = 0; j0 < KPAD; j0 += 32) {
    // ---- stage K tile (32x64, contiguous 4KB) and V tile (64 rows x 64B)
    //      into LDS: 256 16-byte chunks each, spread over 128 threads. ----
    {
      const char* gk = (const char*)(K + (size_t)j0 * DHEAD);
      const char* gv = (const char*)(V + (size_t)j0);
#pragma unroll
      for (int rep = 0; rep < 2; ++rep) {
        int idx = t + rep * 128;                 // 0..255
        async_g2l_b128(lk + idx * 16, gk + (size_t)idx * 16);
        // V: row = idx>>2 (d), 16B chunk (idx&3) within the 64B row
        async_g2l_b128(lv + idx * 16,
                       gv + (size_t)(idx >> 2) * (KPAD * 2) + (idx & 3) * 16);
      }
      asm volatile("s_wait_asynccnt 0" ::: "memory");
      __syncthreads();                           // tiles visible to all waves
    }

    // S = Q K^T for 32 keys (two 16-column tiles), contraction over d=64.
    v8f s0 = {}, s1 = {};
    s0 = wmma16(a0, load_b_t(ldsK, DHEAD, 0, 0, lane), s0);
    s0 = wmma16(a1, load_b_t(ldsK, DHEAD, 0, 32, lane), s0);
    s1 = wmma16(a0, load_b_t(ldsK, DHEAD, 16, 0, lane), s1);
    s1 = wmma16(a1, load_b_t(ldsK, DHEAD, 16, 32, lane), s1);

    float mask0 = (j0 + l < MK) ? 0.f : -1e30f;
    float mask1 = (j0 + 16 + l < MK) ? 0.f : -1e30f;
#pragma unroll
    for (int r = 0; r < 8; ++r) {
      s0[r] = s0[r] * scale + mask0;
      s1[r] = s1[r] * scale + mask1;
    }

    // Online softmax: rows 0..7 live on lanes 0..15, rows 8..15 on lanes
    // 16..31, so xor-shuffles 1/2/4/8 reduce each row across its 16 columns.
#pragma unroll
    for (int r = 0; r < 8; ++r) {
      float mx = fmaxf(s0[r], s1[r]);
#pragma unroll
      for (int mstep = 8; mstep >= 1; mstep >>= 1)
        mx = fmaxf(mx, __shfl_xor(mx, mstep, 32));
      float mnew = fmaxf(m_i[r], mx);
      float alpha = __expf(m_i[r] - mnew);
      float p0 = __expf(s0[r] - mnew);
      float p1 = __expf(s1[r] - mnew);
      s0[r] = p0; s1[r] = p1;
      float sum = p0 + p1;
#pragma unroll
      for (int mstep = 8; mstep >= 1; mstep >>= 1)
        sum += __shfl_xor(sum, mstep, 32);
      l_i[r] = l_i[r] * alpha + sum;
      m_i[r] = mnew;
#pragma unroll
      for (int seg = 0; seg < 4; ++seg) acc[seg][r] *= alpha;
    }

    // C-layout P -> row-major LDS (16x32) -> A-layout fragment (per wave).
    _Float16* pb = &pbuf[wave][0];
#pragma unroll
    for (int r = 0; r < 8; ++r) {
      pb[(hi * 8 + r) * 32 + l]      = (_Float16)s0[r];
      pb[(hi * 8 + r) * 32 + l + 16] = (_Float16)s1[r];
    }
    asm volatile("s_wait_dscnt 0" ::: "memory");
    v16h pa = load_a_f16(pb, 32, 0, 0, lane);

    // O += P (16x32) @ V (32x64): 4 WMMAs over the d-segments, K = 32 keys.
#pragma unroll
    for (int seg = 0; seg < 4; ++seg)
      acc[seg] = wmma16(pa, load_b_t(ldsV, 32, seg * 16, 0, lane), acc[seg]);

    // All LDS reads done before the next iteration overwrites the tiles.
    asm volatile("s_wait_dscnt 0" ::: "memory");
    __syncthreads();
  }

  // Normalize and store to ao [B*N][INNER] (f16), merging heads back.
  int b = bh / HEADS_, h = bh % HEADS_;
#pragma unroll
  for (int seg = 0; seg < 4; ++seg) {
#pragma unroll
    for (int r = 0; r < 8; ++r) {
      int n = row0 + hi * 8 + r;
      ao[((size_t)b * N_ + n) * INNER_ + h * DHEAD + seg * 16 + l] =
          (_Float16)(acc[seg][r] / l_i[r]);
    }
  }
}

// ---------------------------------------------------------------------------
// Kernel 4: out = ao @ Wout + bout  (f32 output)
// ---------------------------------------------------------------------------
__global__ void xattn_proj_out(const _Float16* __restrict__ ao,
                               const _Float16* __restrict__ WoutT,
                               const float* __restrict__ bout,
                               float* __restrict__ out) {
  int wave = threadIdx.x >> 5, lane = threadIdx.x & 31;
  int tile = blockIdx.x * WPB + wave;            // (B*N/16) * (512/16) tiles
  const int tilesN = INNER_ / 16;                // 32
  int row0 = (tile / tilesN) * 16;
  int col0 = (tile % tilesN) * 16;
  v8f c = {};
#pragma unroll 4
  for (int k0 = 0; k0 < INNER_; k0 += 32) {
    v16h a = load_a_f16(ao, INNER_, row0, k0, lane);
    v16h b = load_b_t(WoutT, INNER_, col0, k0, lane);
    c = wmma16(a, b, c);
  }
  int l = lane & 15, hi = lane >> 4;
  int col = col0 + l;
  float bias = bout[col];
#pragma unroll
  for (int r = 0; r < 8; ++r)
    out[(size_t)(row0 + hi * 8 + r) * INNER_ + col] = c[r] + bias;
}

// ---------------------------------------------------------------------------
extern "C" void kernel_launch(void* const* d_in, const int* in_sizes, int n_in,
                              void* d_out, int out_size, void* d_ws,
                              size_t ws_size, hipStream_t stream) {
  const float* x      = (const float*)d_in[0];
  const float* ctx    = (const float*)d_in[1];
  const float* Wq     = (const float*)d_in[2];
  const float* Wkv    = (const float*)d_in[3];
  const float* null_k = (const float*)d_in[4];
  const float* null_v = (const float*)d_in[5];
  const float* Wout   = (const float*)d_in[6];
  const float* bout   = (const float*)d_in[7];
  float* out = (float*)d_out;

  _Float16* ws = (_Float16*)d_ws;
  size_t off = 0;
  _Float16* WqT  = ws + off; off += (size_t)INNER_ * DIMX;        // 262144
  _Float16* WkvT = ws + off; off += (size_t)2 * INNER_ * DIMX;    // 524288
  _Float16* WoutT= ws + off; off += (size_t)INNER_ * INNER_;      // 262144
  _Float16* qh   = ws + off; off += (size_t)B_ * HEADS_ * N_ * DHEAD;
  _Float16* kh   = ws + off; off += (size_t)B_ * HEADS_ * KPAD * DHEAD;
  _Float16* vT   = ws + off; off += (size_t)B_ * HEADS_ * DHEAD * KPAD;
  _Float16* ao   = ws + off; off += (size_t)B_ * N_ * INNER_;

  // Weights -> f16, transposed (column-contiguous for B fragments).
  xattn_w_to_f16T<<<(DIMX * INNER_) / 256, 256, 0, stream>>>(Wq, WqT, DIMX, INNER_);
  xattn_w_to_f16T<<<(DIMX * 2 * INNER_) / 256, 256, 0, stream>>>(Wkv, WkvT, DIMX, 2 * INNER_);
  xattn_w_to_f16T<<<(INNER_ * INNER_) / 256, 256, 0, stream>>>(Wout, WoutT, INNER_, INNER_);

  // Projections.
  xattn_proj_q<<<(B_ * N_ / 16) * (INNER_ / 16) / WPB, WPB * 32, 0, stream>>>(x, WqT, qh);
  xattn_proj_kv<<<(B_ * M_ / 16) * (2 * INNER_ / 16) / WPB, WPB * 32, 0, stream>>>(ctx, WkvT, kh, vT);
  xattn_null_pad<<<B_ * HEADS_, DHEAD, 0, stream>>>(null_k, null_v, kh, vT);

  // Flash attention.
  xattn_attn<<<B_ * HEADS_ * (N_ / 16) / WPB, WPB * 32, 0, stream>>>(qh, kh, vT, ao);

  // Output projection.
  xattn_proj_out<<<(B_ * N_ / 16) * (INNER_ / 16) / WPB, WPB * 32, 0, stream>>>(ao, WoutT, bout, out);
}
